// HyperAttention_29807073034519
// MI455X (gfx1250) — matrix operand order repaired and checked by reading, not compiled, not run on hardware
//
#include <hip/hip_runtime.h>

typedef __attribute__((ext_vector_type(16))) _Float16 v16h;
typedef __attribute__((ext_vector_type(8)))  _Float16 v8h;
typedef __attribute__((ext_vector_type(8)))  float    v8f;
typedef __attribute__((ext_vector_type(4)))  float    v4f;

#define BATCH 2
#define HEADS 16
#define NTOK  8192
#define DIM   64
#define NPROJ 7
#define BLK   256
#define SAMP  256
#define BH    (BATCH*HEADS)
#define BHN   (BH*NTOK)

#define SSTRIDE 36   // floats per Sbuf row (bank-conflict padding, 16B-aligned)
#define PSTRIDE 40   // halves per Pbuf row (bank-conflict padding, 16B-aligned)

// ---------------------------------------------------------------------------
// Kernel 1: LSH hash with gray-code bucket ids.  gray(i) = i ^ (i>>1).
// ---------------------------------------------------------------------------
__global__ __launch_bounds__(256) void hash_kernel(const float* __restrict__ gq,
                                                   const float* __restrict__ gk,
                                                   const float* __restrict__ proj,
                                                   int* __restrict__ qh,
                                                   int* __restrict__ kh) {
  __shared__ float pr[DIM * NPROJ];
  const int t = threadIdx.x;
  for (int i = t; i < DIM * NPROJ; i += 256) pr[i] = proj[i];
  __syncthreads();
  const int row = blockIdx.x * 256 + t;
  const float* x = (blockIdx.y == 0 ? gq : gk) + (size_t)row * DIM;
  float acc[NPROJ];
#pragma unroll
  for (int r = 0; r < NPROJ; ++r) acc[r] = 0.f;
  for (int d = 0; d < DIM; ++d) {
    float xv = x[d];
#pragma unroll
    for (int r = 0; r < NPROJ; ++r) acc[r] += xv * pr[d * NPROJ + r];
  }
  int bin = 0;
#pragma unroll
  for (int r = 0; r < NPROJ; ++r) bin |= (acc[r] > 0.f ? 1 : 0) << r;
  int gray = bin ^ (bin >> 1);
  (blockIdx.y == 0 ? qh : kh)[row] = gray;
}

// ---------------------------------------------------------------------------
// Kernel 2: stable counting sort (128 buckets) per (b,h) row of 8192.
// One 128-thread block; thread t owns contiguous chunk [t*64, t*64+64).
// ---------------------------------------------------------------------------
__global__ __launch_bounds__(128) void sort_kernel(const int* __restrict__ qh,
                                                   const int* __restrict__ kh,
                                                   int* __restrict__ qs,
                                                   int* __restrict__ ks) {
  __shared__ unsigned short hist[128 * 128];   // [bucket][thread]
  __shared__ unsigned int   btot[128];
  __shared__ unsigned int   bstart[128];
  const int bh  = blockIdx.x;
  const int tid = threadIdx.x;
  const int* h  = (blockIdx.y == 0 ? qh : kh) + (size_t)bh * NTOK;
  int*       out = (blockIdx.y == 0 ? qs : ks) + (size_t)bh * NTOK;

  for (int i = 0; i < 128; ++i) hist[i * 128 + tid] = 0;
  __syncthreads();
  const int base = tid * 64;
  for (int e = 0; e < 64; ++e) { int b = h[base + e]; hist[b * 128 + tid]++; }
  __syncthreads();
  {  // thread tid scans bucket row tid over 128 thread-columns -> exclusive prefix
    unsigned run = 0;
    for (int t = 0; t < 128; ++t) {
      unsigned c = hist[tid * 128 + t];
      hist[tid * 128 + t] = (unsigned short)run;
      run += c;
    }
    btot[tid] = run;
  }
  __syncthreads();
  if (tid == 0) {
    unsigned acc = 0;
    for (int b = 0; b < 128; ++b) { bstart[b] = acc; acc += btot[b]; }
  }
  __syncthreads();
  for (int e = 0; e < 64; ++e) {
    int i = base + e;
    int b = h[i];
    unsigned pos = bstart[b] + hist[b * 128 + tid];
    hist[b * 128 + tid]++;
    out[pos] = i;                       // stable: chunk processed in order
  }
}

// ---------------------------------------------------------------------------
// Kernel 3: fused flash attention over 512 keys (256 block + 256 sampled).
// ---------------------------------------------------------------------------
__device__ inline v16h ld16(const _Float16* p0, const _Float16* p1) {
  v8h a = *(const v8h*)p0;
  v8h b = *(const v8h*)p1;
  v16h r;
#pragma unroll
  for (int i = 0; i < 8; ++i) { r[i] = a[i]; r[8 + i] = b[i]; }
  return r;
}

__device__ inline v16h ldQ(const float* p, int half, int dc) {
  // A-layout: lane holds row M; halves 0..7 -> K=8*half+i, halves 8..15 -> K=16+8*half+i
  v16h r;
#pragma unroll
  for (int i = 0; i < 8; ++i) {
    r[i]     = (_Float16)(0.125f * p[dc * 32 + 8 * half + i]);
    r[8 + i] = (_Float16)(0.125f * p[dc * 32 + 16 + 8 * half + i]);
  }
  return r;
}

#define WMMA_F16(A, B, C) \
  __builtin_amdgcn_wmma_f32_16x16x32_f16(false, (A), false, (B), (short)0, (C), false, false)

__global__ __launch_bounds__(256) void fused_attn(const float* __restrict__ gq,
                                                  const float* __restrict__ gk,
                                                  const float* __restrict__ gv,
                                                  const int* __restrict__ samp,
                                                  const int* __restrict__ qsort,
                                                  const int* __restrict__ ksort,
                                                  float* __restrict__ out) {
  __shared__ __attribute__((aligned(16))) _Float16 Ks[512 * 64];           // 64 KB
  __shared__ __attribute__((aligned(16))) _Float16 Vt[64 * 512];           // 64 KB
  __shared__ __attribute__((aligned(16))) float    Sbuf[256 * SSTRIDE];    // 36 KB
  __shared__ __attribute__((aligned(16))) _Float16 Pbuf[256 * PSTRIDE];    // 20 KB
  __shared__ float kbias[512];
  __shared__ float Abuf[256];
  __shared__ float Lbuf[256];
  __shared__ int   qmap[256];

  const int qb  = blockIdx.x;     // sorted-order query block (== key block id for mask)
  const int bh  = blockIdx.y;
  const int tid = threadIdx.x;
  const int w    = tid >> 5;
  const int ln   = tid & 31;
  const int lnm  = ln & 15;
  const int half = ln >> 4;

  // ---- stage K, V(T), biases, q row map -----------------------------------
  qmap[tid] = qsort[(size_t)bh * NTOK + qb * BLK + tid];
  {
    // block-diagonal key row tid
    int js0 = qb * BLK + tid;
    int o0  = ksort[(size_t)bh * NTOK + js0];
    const float* kp = gk + ((size_t)bh * NTOK + o0) * DIM;
    const float* vp = gv + ((size_t)bh * NTOK + o0) * DIM;
#pragma unroll 8
    for (int d = 0; d < DIM; ++d) {
      Ks[tid * DIM + d]  = (_Float16)kp[d];
      Vt[d * 512 + tid]  = (_Float16)vp[d];
    }
    kbias[tid] = 0.f;
    // sampled key row tid
    int js1 = samp[(size_t)bh * SAMP + tid];     // index into sorted key order
    int o1  = ksort[(size_t)bh * NTOK + js1];
    kp = gk + ((size_t)bh * NTOK + o1) * DIM;
    vp = gv + ((size_t)bh * NTOK + o1) * DIM;
#pragma unroll 8
    for (int d = 0; d < DIM; ++d) {
      Ks[(256 + tid) * DIM + d] = (_Float16)kp[d];
      Vt[d * 512 + 256 + tid]   = (_Float16)vp[d];
    }
    // merged-softmax column bias: +log(N/SAMPLE), or -inf if sample is in this block
    kbias[256 + tid] = ((js1 >> 8) == qb) ? -1e30f : 3.4657359028f;
  }
  __syncthreads();

  // ---- Q tiles into registers (A-layout), pre-scaled by D^-0.5 ------------
  v16h aQ[2][2];
#pragma unroll
  for (int mt = 0; mt < 2; ++mt) {
    int row = w * 32 + mt * 16 + lnm;
    const float* qp = gq + ((size_t)bh * NTOK + qmap[row]) * DIM;
#pragma unroll
    for (int dc = 0; dc < 2; ++dc) aQ[mt][dc] = ldQ(qp, half, dc);
  }

  // ---- online-softmax state ------------------------------------------------
  // O accumulators live in C-layout lanes; running (m,l) live in the
  // row-owner lane (lane ln owns wave-local query row ln).
  v8f  O[2][4];
#pragma unroll
  for (int mt = 0; mt < 2; ++mt)
#pragma unroll
    for (int nd = 0; nd < 4; ++nd) { v8f z = {}; O[mt][nd] = z; }
  float mrow = -1e30f;
  float lrow = 0.f;

  // ---- main loop: 16 steps of 32 keys -------------------------------------
  for (int step = 0; step < 16; ++step) {
    const int kb = step * 32;
    v16h bk[2][2];
#pragma unroll
    for (int nt = 0; nt < 2; ++nt)
#pragma unroll
      for (int dc = 0; dc < 2; ++dc) {
        const _Float16* p = &Ks[(kb + nt * 16 + lnm) * DIM + dc * 32 + 8 * half];
        bk[nt][dc] = ld16(p, p + 16);
      }
    const float bias0 = kbias[kb + lnm];
    const float bias1 = kbias[kb + 16 + lnm];

    // GEMM1; scatter biased scores C-layout -> row-major Sbuf
#pragma unroll
    for (int mt = 0; mt < 2; ++mt) {
      v8f s0 = {}, s1 = {};
#pragma unroll
      for (int dc = 0; dc < 2; ++dc) {
        s0 = WMMA_F16(aQ[mt][dc], bk[0][dc], s0);
        s1 = WMMA_F16(aQ[mt][dc], bk[1][dc], s1);
      }
#pragma unroll
      for (int g = 0; g < 8; ++g) {
        int prow = w * 32 + mt * 16 + g + 8 * half;
        Sbuf[prow * SSTRIDE + lnm]      = s0[g] + bias0;
        Sbuf[prow * SSTRIDE + 16 + lnm] = s1[g] + bias1;
      }
    }
    asm volatile("s_wait_dscnt 0" ::: "memory");   // same-wave LDS handoff

    // Row-owner softmax pass: lane ln owns wave row ln (no cross-lane ops)
    {
      const float* srow = &Sbuf[(w * 32 + ln) * SSTRIDE];
      float sv[32];
#pragma unroll
      for (int c = 0; c < 8; ++c) {
        v4f t = *(const v4f*)(srow + 4 * c);
#pragma unroll
        for (int j = 0; j < 4; ++j) sv[4 * c + j] = t[j];
      }
      float mloc = sv[0];
#pragma unroll
      for (int i = 1; i < 32; ++i) mloc = fmaxf(mloc, sv[i]);
      float mn    = fmaxf(mrow, mloc);
      float alpha = __expf(mrow - mn);
      mrow = mn;
      float sum = 0.f;
#pragma unroll
      for (int i = 0; i < 32; ++i) {
        float p = __expf(sv[i] - mn);
        sv[i] = p;
        sum += p;
      }
      lrow = lrow * alpha + sum;
      Abuf[w * 32 + ln] = alpha;
      _Float16* prow = &Pbuf[(w * 32 + ln) * PSTRIDE];
#pragma unroll
      for (int c = 0; c < 4; ++c) {
        v8h ph;
#pragma unroll
        for (int j = 0; j < 8; ++j) ph[j] = (_Float16)sv[c * 8 + j];
        *(v8h*)(prow + c * 8) = ph;
      }
    }
    asm volatile("s_wait_dscnt 0" ::: "memory");   // same-wave LDS handoff

    // Rescale O by per-row alpha (broadcast LDS reads into C-layout lanes)
    float alr[2][8];
#pragma unroll
    for (int mt = 0; mt < 2; ++mt)
#pragma unroll
      for (int g = 0; g < 8; ++g) alr[mt][g] = Abuf[w * 32 + mt * 16 + g + 8 * half];
#pragma unroll
    for (int mt = 0; mt < 2; ++mt)
#pragma unroll
      for (int nd = 0; nd < 4; ++nd)
#pragma unroll
        for (int g = 0; g < 8; ++g) O[mt][nd][g] *= alr[mt][g];

    // GEMM2: O += P * V
    v16h bv[4];
#pragma unroll
    for (int nd = 0; nd < 4; ++nd) {
      const _Float16* p = &Vt[(nd * 16 + lnm) * 512 + kb + 8 * half];
      bv[nd] = ld16(p, p + 16);
    }
    v16h aP[2];
#pragma unroll
    for (int mt = 0; mt < 2; ++mt) {
      const _Float16* p = &Pbuf[(w * 32 + mt * 16 + lnm) * PSTRIDE];
      aP[mt] = ld16(p + 8 * half, p + 16 + 8 * half);
    }
#pragma unroll
    for (int mt = 0; mt < 2; ++mt)
#pragma unroll
      for (int nd = 0; nd < 4; ++nd) O[mt][nd] = WMMA_F16(aP[mt], bv[nd], O[mt][nd]);
    asm volatile("" ::: "memory");
  }

  // ---- epilogue: publish l, normalize, scatter back through q_sort --------
  Lbuf[w * 32 + ln] = lrow;
  asm volatile("s_wait_dscnt 0" ::: "memory");
  float li[2][8];
#pragma unroll
  for (int mt = 0; mt < 2; ++mt)
#pragma unroll
    for (int g = 0; g < 8; ++g) li[mt][g] = Lbuf[w * 32 + mt * 16 + g + 8 * half];
#pragma unroll
  for (int mt = 0; mt < 2; ++mt)
#pragma unroll
    for (int nd = 0; nd < 4; ++nd)
#pragma unroll
      for (int g = 0; g < 8; ++g) {
        int row = w * 32 + mt * 16 + g + 8 * half;
        int col = nd * 16 + lnm;
        float val = O[mt][nd][g] / li[mt][g];
        out[((size_t)bh * NTOK + qmap[row]) * DIM + col] = val;
      }
}

// ---------------------------------------------------------------------------
extern "C" void kernel_launch(void* const* d_in, const int* in_sizes, int n_in,
                              void* d_out, int out_size, void* d_ws, size_t ws_size,
                              hipStream_t stream) {
  const float* q    = (const float*)d_in[0];
  const float* k    = (const float*)d_in[1];
  const float* v    = (const float*)d_in[2];
  const float* proj = (const float*)d_in[3];
  const int*   samp = (const int*)d_in[4];
  float* out = (float*)d_out;

  int* qh = (int*)d_ws;
  int* kh = qh + BHN;
  int* qs = kh + BHN;
  int* ks = qs + BHN;

  hash_kernel<<<dim3(BHN / 256, 2), 256, 0, stream>>>(q, k, proj, qh, kh);
  sort_kernel<<<dim3(BH, 2), 128, 0, stream>>>(qh, kh, qs, ks);
  fused_attn<<<dim3(NTOK / BLK, BH), 256, 0, stream>>>(q, k, v, samp, qs, ks, out);
}
